// G_TCN_32066225832587
// MI455X (gfx1250) — matrix operand order repaired and checked by reading, not compiled, not run on hardware
//
#include <hip/hip_runtime.h>
#include <hip/hip_bf16.h>
#include <stdint.h>
#include <stddef.h>

// ---------------------------------------------------------------------------
// Problem constants (from reference)
// ---------------------------------------------------------------------------
#define T_  8
#define NN  10000
#define EE  100000
#define EPSBN 1e-5f

typedef __attribute__((ext_vector_type(16))) __bf16 v16bf;
typedef __attribute__((ext_vector_type(8)))  __bf16 v8bf;
typedef __attribute__((ext_vector_type(2)))  __bf16 v2bf;
typedef __attribute__((ext_vector_type(8)))  float  v8f;
typedef __attribute__((ext_vector_type(4)))  float  v4f;
typedef __attribute__((ext_vector_type(4)))  unsigned v4u;
typedef __attribute__((ext_vector_type(8)))  int      v8i;
typedef __attribute__((ext_vector_type(4)))  int      v4i;

#define HAVE_TDM (__has_builtin(__builtin_amdgcn_tensor_load_to_lds) && \
                  __has_builtin(__builtin_amdgcn_s_wait_tensorcnt))

__device__ __forceinline__ __bf16 f2bf(float f) {       // RNE f32 -> bf16
  union { float f; unsigned u; } x; x.f = f;
  unsigned r = x.u + 0x7fffu + ((x.u >> 16) & 1u);
  union { unsigned short s; __bf16 b; } y; y.s = (unsigned short)(r >> 16);
  return y.b;
}
// Pack 16 f32 (4x v4f) into a WMMA bf16 fragment (RNE).
__device__ __forceinline__ v16bf pack16(v4f f0, v4f f1, v4f f2, v4f f3) {
  v16bf a;
#if __has_builtin(__builtin_amdgcn_cvt_pk_bf16_f32)
  v2bf p;
  p = __builtin_amdgcn_cvt_pk_bf16_f32(f0[0], f0[1]); a[0] = p[0];  a[1] = p[1];
  p = __builtin_amdgcn_cvt_pk_bf16_f32(f0[2], f0[3]); a[2] = p[0];  a[3] = p[1];
  p = __builtin_amdgcn_cvt_pk_bf16_f32(f1[0], f1[1]); a[4] = p[0];  a[5] = p[1];
  p = __builtin_amdgcn_cvt_pk_bf16_f32(f1[2], f1[3]); a[6] = p[0];  a[7] = p[1];
  p = __builtin_amdgcn_cvt_pk_bf16_f32(f2[0], f2[1]); a[8] = p[0];  a[9] = p[1];
  p = __builtin_amdgcn_cvt_pk_bf16_f32(f2[2], f2[3]); a[10] = p[0]; a[11] = p[1];
  p = __builtin_amdgcn_cvt_pk_bf16_f32(f3[0], f3[1]); a[12] = p[0]; a[13] = p[1];
  p = __builtin_amdgcn_cvt_pk_bf16_f32(f3[2], f3[3]); a[14] = p[0]; a[15] = p[1];
#else
  #pragma unroll
  for (int p4 = 0; p4 < 4; ++p4) {
    a[p4]      = f2bf(f0[p4]);
    a[4 + p4]  = f2bf(f1[p4]);
    a[8 + p4]  = f2bf(f2[p4]);
    a[12 + p4] = f2bf(f3[p4]);
  }
#endif
  return a;
}
// monotonic float<->uint encoding for atomicMax-based segment max
__device__ __forceinline__ unsigned fenc(float f) {
  unsigned u = __float_as_uint(f);
  return (u & 0x80000000u) ? ~u : (u | 0x80000000u);
}
__device__ __forceinline__ float fdec(unsigned e) {
  unsigned u = (e & 0x80000000u) ? (e & 0x7fffffffu) : ~e;
  return __uint_as_float(u);
}
#define ENC_NEG_INF 0x007FFFFFu   // fenc(-inf)

// Stage `bytes` (16B-multiple) of packed weights into LDS offset 0.
// Prefer the Tensor Data Mover (1-D descriptor DMA, one issue per block),
// falling back to a cooperative vector copy.
__device__ __forceinline__ void stage_weights(const __bf16* __restrict__ WtP,
                                              char* smem, int bytes, int wid)
{
#if HAVE_TDM
  if (wid == 0) {                       // one wave issues the TDM descriptor
    const unsigned long long ga = (unsigned long long)(const void*)WtP;
    const unsigned ndw = (unsigned)bytes >> 2;     // dwords (fits 16-bit tile dim)
    v4u g0;
    g0[0] = 1u;                                    // count=1 (valid), user mode
    g0[1] = 0u;                                    // lds_addr = 0
    g0[2] = (unsigned)(ga & 0xFFFFFFFFu);          // global_addr[31:0]
    g0[3] = (unsigned)((ga >> 32) & 0x01FFFFFFu) | (2u << 30);  // addr[56:32], type=2
    v8i g1;
    g1[0] = (int)(2u << 16);                       // data_size = 4B, no flags
    g1[1] = (int)((ndw & 0xFFFFu) << 16);          // tensor_dim0 lo16 @ [63:48]
    g1[2] = (int)((ndw >> 16) & 0xFFFFu);          // tensor_dim0 hi16 @ [79:64]
    g1[3] = (int)((ndw & 0xFFFFu) << 16);          // tile_dim0 @ [127:112]
    g1[4] = 0;                                     // tile_dim1/2 = 0 (1-D tile)
    g1[5] = (int)ndw;                              // tensor_dim0_stride lo32
    g1[6] = 0; g1[7] = 0;
    v4i gz = {0, 0, 0, 0};
    v8i gz8 = {0, 0, 0, 0, 0, 0, 0, 0};
    __builtin_amdgcn_tensor_load_to_lds(g0, g1, gz, gz, gz8, 0);
    __builtin_amdgcn_s_wait_tensorcnt(0);
  }
#else
  (void)wid;
  const int words = bytes >> 4;
  const uint4* s = (const uint4*)WtP;
  uint4* d = (uint4*)smem;
  for (int i = threadIdx.x; i < words; i += 256) d[i] = s[i];
#endif
  __syncthreads();
}

// ---------------------------------------------------------------------------
// Generic multi-tap WMMA GEMM, compile-time tiling:
//   NT   = number of 16-wide N tiles (Npad/16), fully unrolled
//   MR   = 16-row strips per wave
//   KPER = K per tap, unrolled in 32-steps
//   AB   = A operand stored as bf16 (direct fragment loads, no convert VALU)
//   OB   = output stored as bf16 (feeds the next GEMM's A)
// Out[t, r, n] = epi( sum_j  A_j[t+ts_j, rowmap_j(r), :KPER] @ Wt_j[:, n] )
// Taps with t+ts_j < 0 contribute zero (causal conv padding).
// Wt packed bf16 as [tap][ntile][n%16][KPER] by k_pack_w (staged in LDS via TDM).
// Epilogue: y = acc*scale[n]+bias[n]; if relu: y=max(y,0); y += Res[t,r,n] (f32).
// ---------------------------------------------------------------------------
template<int NT, int MR, int KPER, bool AB, bool OB>
__global__ __launch_bounds__(256) void k_gemm(
    const void* __restrict__ A0, const void* __restrict__ A1, const void* __restrict__ A2,
    const int* __restrict__ idx0, const int* __restrict__ idx1, const int* __restrict__ idx2,
    int Ma0, int Ma1, int Ma2, int ts0, int ts1, int ts2,
    int nTap,
    const __bf16* __restrict__ WtP,
    const float* __restrict__ scale, const float* __restrict__ bias,
    const float* __restrict__ Res,
    void* __restrict__ Out,
    int Mt, int Nout, int doRelu)
{
  constexpr int NPAD = NT * 16;
  extern __shared__ __align__(16) char smem_raw[];
  __bf16* sW = (__bf16*)smem_raw;

  const int wid   = threadIdx.x >> 5;
  stage_weights(WtP, smem_raw, nTap * NPAD * KPER * 2, wid);

  const int lane  = threadIdx.x & 31;
  const int lrow  = lane & 15;
  const int lhalf = lane >> 4;
  const int t     = blockIdx.y;
  const int row0  = blockIdx.x * (128 * MR) + wid * (16 * MR);
  if (row0 >= Mt) return;                 // wave-uniform

  v8f acc[MR][NT];
  #pragma unroll
  for (int s = 0; s < MR; ++s)
    #pragma unroll
    for (int i = 0; i < NT; ++i)
      #pragma unroll
      for (int v = 0; v < 8; ++v) acc[s][i][v] = 0.f;

  for (int j = 0; j < nTap; ++j) {
    const void* A   = (j == 0) ? A0 : ((j == 1) ? A1 : A2);
    const int* idx  = (j == 0) ? idx0 : ((j == 1) ? idx1 : idx2);
    const int Ma    = (j == 0) ? Ma0 : ((j == 1) ? Ma1 : Ma2);
    const int ts    = (j == 0) ? ts0 : ((j == 1) ? ts1 : ts2);
    const int tt = t + ts;
    if (tt < 0) continue;                 // causal zero padding

    size_t rofs[MR];
    #pragma unroll
    for (int s = 0; s < MR; ++s) {
      int rs = row0 + s * 16 + lrow;
      if (rs > Mt - 1) rs = Mt - 1;       // clamp partial strip (stores guarded)
      const int ra = idx ? idx[rs] : rs;
      rofs[s] = ((size_t)tt * (size_t)Ma + (size_t)ra) * (size_t)KPER;
    }
    const __bf16* wbase = sW + (size_t)j * NPAD * KPER;

    #pragma unroll
    for (int kb = 0; kb < KPER; kb += 32) {
      // A fragment: lane holds row M=lrow, K = c0..c0+7 and c0+16..c0+23
      const int c0 = kb + lhalf * 8;
      v16bf a[MR];
      #pragma unroll
      for (int s = 0; s < MR; ++s) {
        if (AB) {
          const __bf16* ap = (const __bf16*)A + rofs[s];
          v8bf lo = *(const v8bf*)(ap + c0);
          v8bf hi = *(const v8bf*)(ap + c0 + 16);
          a[s] = __builtin_shufflevector(lo, hi, 0, 1, 2, 3, 4, 5, 6, 7,
                                         8, 9, 10, 11, 12, 13, 14, 15);
        } else {
          const float* ap = (const float*)A + rofs[s];
          v4f f0 = *(const v4f*)(ap + c0);
          v4f f1 = *(const v4f*)(ap + c0 + 4);
          v4f f2 = *(const v4f*)(ap + c0 + 16);
          v4f f3 = *(const v4f*)(ap + c0 + 20);
          a[s] = pack16(f0, f1, f2, f3);
        }
      }
      // preload all B fragments, then a back-to-back WMMA chain
      v16bf bfr[NT];
      #pragma unroll
      for (int nt = 0; nt < NT; ++nt) {
        const __bf16* bp = wbase + (size_t)(nt * 16 + lrow) * KPER + kb + lhalf * 8;
        v8bf blo = *(const v8bf*)(bp);
        v8bf bhi = *(const v8bf*)(bp + 16);
        bfr[nt] = __builtin_shufflevector(blo, bhi, 0, 1, 2, 3, 4, 5, 6, 7,
                                          8, 9, 10, 11, 12, 13, 14, 15);
      }
      #pragma unroll
      for (int nt = 0; nt < NT; ++nt)
        #pragma unroll
        for (int s = 0; s < MR; ++s)
          acc[s][nt] = __builtin_amdgcn_wmma_f32_16x16x32_bf16(
              false, a[s], false, bfr[nt], (short)0, acc[s][nt], false, false);
    }
  }

  #pragma unroll
  for (int nt = 0; nt < NT; ++nt) {
    const int col = nt * 16 + lrow;
    if (col < Nout) {
      const float sc = scale ? scale[col] : 1.f;
      const float bs = bias ? bias[col] : 0.f;
      #pragma unroll
      for (int s = 0; s < MR; ++s) {
        #pragma unroll
        for (int v = 0; v < 8; ++v) {
          const int rr = row0 + s * 16 + lhalf * 8 + v;   // D: M = v + 8*(lane/16)
          if (rr < Mt) {
            float y = acc[s][nt][v] * sc + bs;
            if (doRelu) y = fmaxf(y, 0.f);
            const size_t o = ((size_t)t * Mt + rr) * (size_t)Nout + col;
            if (Res) y += Res[o];          // residual after relu (inputs >= 0)
            if (OB) ((__bf16*)Out)[o] = f2bf(y);
            else    ((float*)Out)[o]  = y;
          }
        }
      }
    }
  }
}

// ---------------------------------------------------------------------------
// Weight packing: fp32 -> bf16 in WMMA-B-fragment-friendly layout
// mode 0: W is [nTap*KPER][Nout] row-major (linear layers / cls)
// mode 1: W is TCN conv weight [co][ci][ktap] (c,c,3): tap j, kk=ci, n=co
// ---------------------------------------------------------------------------
__global__ void k_pack_w(const float* __restrict__ Wsrc, __bf16* __restrict__ dst,
                         int KPER, int Nout, int Npad, int nTap, int mode)
{
  int i = blockIdx.x * 256 + threadIdx.x;
  int total = nTap * Npad * KPER;
  if (i >= total) return;
  int kk  = i % KPER;
  int n16 = (i / KPER) & 15;
  int nt  = (i / (KPER * 16)) % (Npad >> 4);
  int j   = i / (KPER * Npad);
  int n = nt * 16 + n16;
  float v = 0.f;
  if (n < Nout) {
    if (mode == 0) v = Wsrc[(size_t)(j * KPER + kk) * Nout + n];
    else           v = Wsrc[((size_t)n * KPER + kk) * nTap + j];
  }
  dst[i] = f2bf(v);
}

// fold conv-bias + eval-BatchNorm into per-channel scale/bias
__global__ void k_fold(const float* g, const float* bb, const float* bconv,
                       float* scale, float* bias, int C)
{
  int i = threadIdx.x;
  if (i < C) {
    float s = g[i] * rsqrtf(1.f + EPSBN);
    scale[i] = s;
    bias[i]  = bconv[i] * s + bb[i];
  }
}

// v[h][j] = sum_c Wm[j][h*64+c] * att[h][c]   (Wm is [64][128], att [2][64])
__global__ void k_attvec(const float* __restrict__ Wm, const float* __restrict__ att,
                         float* __restrict__ v)
{
  int i = threadIdx.x;               // 128 threads
  if (i < 128) {
    int h = i >> 6, j = i & 63;
    float s = 0.f;
    #pragma unroll
    for (int c = 0; c < 64; ++c) s += Wm[j * 128 + h * 64 + c] * att[h * 64 + c];
    v[i] = s;
  }
}

__global__ void k_fill_f(float* p, float v, size_t n) {
  size_t i = (size_t)blockIdx.x * 256 + threadIdx.x;
  size_t st = (size_t)gridDim.x * 256;
  for (; i < n; i += st) p[i] = v;
}
__global__ void k_fill_u(unsigned* p, unsigned v, size_t n) {
  size_t i = (size_t)blockIdx.x * 256 + threadIdx.x;
  size_t st = (size_t)gridDim.x * 256;
  for (; i < n; i += st) p[i] = v;
}

__global__ void k_deg(const int* __restrict__ dst, float* deg) {
  int e = blockIdx.x * 256 + threadIdx.x;
  if (e < EE) atomicAdd(&deg[dst[e]], 1.f);
}
__global__ void k_loopsum(const float* __restrict__ ea, const int* __restrict__ dst,
                          float* __restrict__ loop) {
  size_t i = (size_t)blockIdx.x * 256 + threadIdx.x;
  if (i >= (size_t)T_ * EE * 64) return;
  size_t me = i >> 6; int c = (int)(i & 63);
  int t = (int)(me / EE), e = (int)(me % EE);
  atomicAdd(&loop[((size_t)t * NN + dst[e]) * 64 + c], ea[me * 64 + c]);
}
__global__ void k_loopdiv(float* loop, const float* deg) {
  size_t i = (size_t)blockIdx.x * 256 + threadIdx.x;
  if (i >= (size_t)T_ * NN * 64) return;
  int n = (int)((i >> 6) % NN);
  loop[i] /= fmaxf(deg[n], 1.f);
}

// a_src/a_dst from node hidden via folded attention vectors
__global__ void k_nodeatt(const float* __restrict__ H, const float* __restrict__ vs,
                          const float* __restrict__ vd, float* as, float* ad, size_t M)
{
  size_t m = (size_t)blockIdx.x * 256 + threadIdx.x;
  if (m >= M) return;
  const float* r = H + m * 64;
  float s0 = 0, s1 = 0, d0 = 0, d1 = 0;
  #pragma unroll
  for (int c = 0; c < 64; ++c) {
    float x = r[c];
    s0 += x * vs[c]; s1 += x * vs[64 + c];
    d0 += x * vd[c]; d1 += x * vd[64 + c];
  }
  as[m * 2] = s0; as[m * 2 + 1] = s1;
  ad[m * 2] = d0; ad[m * 2 + 1] = d1;
}
__global__ void k_dot2(const float* __restrict__ X, const float* __restrict__ v,
                       float* __restrict__ out, size_t M)
{
  size_t m = (size_t)blockIdx.x * 256 + threadIdx.x;
  if (m >= M) return;
  const float* r = X + m * 64;
  float s0 = 0, s1 = 0;
  #pragma unroll
  for (int c = 0; c < 64; ++c) { float x = r[c]; s0 += x * v[c]; s1 += x * v[64 + c]; }
  out[m * 2] = s0; out[m * 2 + 1] = s1;
}

__global__ void k_logit_e(const float* as, const float* ad, const float* ae,
                          const int* __restrict__ src, const int* __restrict__ dst,
                          float* raw, unsigned* mmax)
{
  size_t i = (size_t)blockIdx.x * 256 + threadIdx.x;
  if (i >= (size_t)T_ * EE * 2) return;
  size_t me = i >> 1; int h = (int)(i & 1);
  int t = (int)(me / EE), e = (int)(me % EE);
  size_t ns = (size_t)t * NN + src[e], nd = (size_t)t * NN + dst[e];
  float z = as[ns * 2 + h] + ad[nd * 2 + h] + ae[i];
  z = (z > 0.f) ? z : 0.2f * z;
  raw[i] = z;
  atomicMax(mmax + nd * 2 + h, fenc(z));
}
__global__ void k_logit_n(const float* as, const float* ad, const float* aen,
                          float* raw, unsigned* mmax)
{
  size_t i = (size_t)blockIdx.x * 256 + threadIdx.x;
  if (i >= (size_t)T_ * NN * 2) return;
  float z = as[i] + ad[i] + aen[i];
  z = (z > 0.f) ? z : 0.2f * z;
  raw[i] = z;
  atomicMax(mmax + i, fenc(z));
}
__global__ void k_exps_e(float* raw, const unsigned* mmax, float* ssum,
                         const int* __restrict__ dst)
{
  size_t i = (size_t)blockIdx.x * 256 + threadIdx.x;
  if (i >= (size_t)T_ * EE * 2) return;
  size_t me = i >> 1; int h = (int)(i & 1);
  int t = (int)(me / EE), e = (int)(me % EE);
  size_t nd = (size_t)t * NN + dst[e];
  float ex = expf(raw[i] - fdec(mmax[nd * 2 + h]));
  raw[i] = ex;
  atomicAdd(ssum + nd * 2 + h, ex);
}
__global__ void k_exps_n(float* raw, const unsigned* mmax, float* ssum)
{
  size_t i = (size_t)blockIdx.x * 256 + threadIdx.x;
  if (i >= (size_t)T_ * NN * 2) return;
  float ex = expf(raw[i] - fdec(mmax[i]));
  raw[i] = ex;
  atomicAdd(ssum + i, ex);
}
__global__ void k_scat_e(const float* __restrict__ alp, const float* __restrict__ ssum,
                         const float* __restrict__ XL, const int* __restrict__ src,
                         const int* __restrict__ dst, float* __restrict__ OUT)
{
  size_t i = (size_t)blockIdx.x * 256 + threadIdx.x;
  if (i >= (size_t)T_ * EE * 64) return;
  size_t me = i >> 6; int c = (int)(i & 63);
  int t = (int)(me / EE), e = (int)(me % EE);
  size_t ns = (size_t)t * NN + src[e], nd = (size_t)t * NN + dst[e];
  #pragma unroll
  for (int h = 0; h < 2; ++h) {
    float a = alp[me * 2 + h] / (ssum[nd * 2 + h] + 1e-16f);
    atomicAdd(&OUT[nd * 128 + h * 64 + c], a * XL[ns * 128 + h * 64 + c]);
  }
}
__global__ void k_scat_n(const float* __restrict__ alp, const float* __restrict__ ssum,
                         const float* __restrict__ XL, float* __restrict__ OUT)
{
  size_t i = (size_t)blockIdx.x * 256 + threadIdx.x;
  if (i >= (size_t)T_ * NN * 64) return;
  size_t m = i >> 6; int c = (int)(i & 63);
  #pragma unroll
  for (int h = 0; h < 2; ++h) {
    float a = alp[m * 2 + h] / (ssum[m * 2 + h] + 1e-16f);
    atomicAdd(&OUT[m * 128 + h * 64 + c], a * XL[m * 128 + h * 64 + c]);
  }
}
__global__ void k_gat_final(const float* __restrict__ OUT, const float* __restrict__ bgat,
                            const float* __restrict__ g, const float* __restrict__ b,
                            float* __restrict__ H)
{
  size_t i = (size_t)blockIdx.x * 256 + threadIdx.x;
  if (i >= (size_t)T_ * NN * 64) return;
  size_t m = i >> 6; int c = (int)(i & 63);
  float mean = 0.5f * (OUT[m * 128 + c] + OUT[m * 128 + 64 + c]);
  float y = (mean + bgat[c]) * (g[c] * rsqrtf(1.f + EPSBN)) + b[c];
  H[i] = fmaxf(y, 0.f);
}

// ---------------------------------------------------------------------------
// Host side
// ---------------------------------------------------------------------------
// Input index map: top-level setup_inputs() insertion order, params pytree
// flattened jax-style (dict keys sorted, lists in order).
enum {
  IN_X = 0, IN_EATTR, IN_EIDX,
  P_BN1_B, P_BN1_G, P_BN2_B, P_BN2_G,
  P_CLS1_W, P_CLS1_B, P_CLS2_W, P_CLS2_B,
  P_EFC1_W, P_EFC1_B, P_EFC2_W, P_EFC2_B,
  P_EPRE1_W, P_EPRE1_B, P_EPRE2_W, P_EPRE2_B,
  P_G1_WE, P_G1_WL, P_G1_ADST, P_G1_AEDGE, P_G1_ASRC, P_G1_BIAS,
  P_G2_WE, P_G2_WL, P_G2_ADST, P_G2_AEDGE, P_G2_ASRC, P_G2_BIAS,
  P_NFC1_W, P_NFC1_B, P_NFC2_W, P_NFC2_B,
  TE0_B1, TE0_B2, TE0_BN1B, TE0_BN1G, TE0_BN2B, TE0_BN2G, TE0_W1, TE0_W2,
  TE1_B1, TE1_B2, TE1_BN1B, TE1_BN1G, TE1_BN2B, TE1_BN2G, TE1_W1, TE1_W2,
  TN0_B1, TN0_B2, TN0_BN1B, TN0_BN1G, TN0_BN2B, TN0_BN2G, TN0_W1, TN0_W2,
  TN1_B1, TN1_B2, TN1_BN1B, TN1_BN1G, TN1_BN2B, TN1_BN2G, TN1_W1, TN1_W2,
};

static const size_t SZ_E    = (size_t)T_ * EE * 64;     // 51.2M floats
static const size_t SZ_N64  = (size_t)T_ * NN * 64;
static const size_t SZ_N128 = (size_t)T_ * NN * 128;
static const size_t SZ_E2   = (size_t)T_ * EE * 2;
static const size_t SZ_N2   = (size_t)T_ * NN * 2;

// packed-weight bf16 offsets (elements)
static const size_t p_nfc1 = 0, p_nfc2 = 8192, p_epre1 = 12288, p_epre2 = 16384;
static const size_t p_Wl1 = 20480, p_Wl2 = 28672, p_efc1 = 36864, p_efc2 = 40960;
static const size_t p_tE_b0w1 = 45056,  p_tE_b0w2 = 57344;
static const size_t p_tE_b1w1 = 69632,  p_tE_b1w2 = 81920;
static const size_t p_tN_b0w1 = 94208,  p_tN_b0w2 = 106496;
static const size_t p_tN_b1w1 = 118784, p_tN_b1w2 = 131072;
static const size_t p_cls1 = 143360, p_cls2 = 167936;
static const size_t PB_TOTAL = 171008;                  // padded

static inline int cdiv(size_t a, int b) { return (int)((a + b - 1) / b); }

struct GP {
  const void *A0, *A1, *A2;
  const int *i0, *i1, *i2;
  int Ma0, Ma1, Ma2, ts0, ts1, ts2, nTap, KPER;
  const __bf16* W;
  const float *sc, *bi, *res;
  void* out;
  int Mt, Nout, Npad, relu, gT, abf, obf;
};
template<int NT, int MR, int KPER, bool AB, bool OB>
static inline void gcall(hipStream_t s, const GP& p) {
  dim3 g(cdiv(p.Mt, 128 * MR), p.gT);
  size_t sh = (size_t)p.nTap * (NT * 16) * KPER * 2;
  k_gemm<NT, MR, KPER, AB, OB><<<g, dim3(256), sh, s>>>(
      p.A0, p.A1, p.A2, p.i0, p.i1, p.i2, p.Ma0, p.Ma1, p.Ma2,
      p.ts0, p.ts1, p.ts2, p.nTap, p.W, p.sc, p.bi, p.res, p.out,
      p.Mt, p.Nout, p.relu);
}
static void run_gemm(hipStream_t s, const GP& p) {
  if      (p.Npad == 128 && p.KPER == 64  && !p.abf && !p.obf) gcall<8, 1, 64, false, false>(s, p);
  else if (p.Npad == 128 && p.KPER == 64  && !p.abf &&  p.obf) gcall<8, 1, 64, false, true>(s, p);
  else if (p.Npad == 64  && p.KPER == 128 && !p.abf &&  p.obf) gcall<4, 2, 128, false, true>(s, p);
  else if (p.Npad == 64  && p.KPER == 64  &&  p.abf && !p.obf) gcall<4, 2, 64, true, false>(s, p);
  else if (p.Npad == 64  && p.KPER == 64  && !p.abf &&  p.obf) gcall<4, 2, 64, false, true>(s, p);
  else if (p.Npad == 64  && p.KPER == 64  && !p.abf && !p.obf) gcall<4, 2, 64, false, false>(s, p);
  else                                                          gcall<1, 2, 128, true, false>(s, p); // cls2
}

extern "C" void kernel_launch(void* const* d_in, const int* in_sizes, int n_in,
                              void* d_out, int out_size, void* d_ws, size_t ws_size,
                              hipStream_t stream)
{
  (void)in_sizes; (void)n_in; (void)out_size;
  const float* X   = (const float*)d_in[IN_X];
  const float* EAT = (const float*)d_in[IN_EATTR];
  const int*   SRC = (const int*)d_in[IN_EIDX];
  const int*   DST = SRC + EE;
  #define PF(i) ((const float*)d_in[(i)])

  // ---- workspace carve-up (floats) ----
  float* Wb = (float*)d_ws;
  size_t o = 0;
  float*    EA0  = Wb + o; o += SZ_E;        // f32 ea; later reused as bf16 H1
  float*    EA1  = Wb + o; o += SZ_E;        // f32
  float*    EA2  = Wb + o; o += SZ_E;        // f32
  __bf16*   EB0  = (__bf16*)(Wb + o); o += SZ_E / 2;   // bf16 edge GEMM temp
  float*    HN0  = Wb + o; o += SZ_N64;
  float*    HN1  = Wb + o; o += SZ_N64;
  float*    HN2  = Wb + o; o += SZ_N64;
  __bf16*   HB0  = (__bf16*)(Wb + o); o += SZ_N64 / 2; // bf16 node GEMM temp
  float*    LOOP = Wb + o; o += SZ_N64;
  float*    XL   = Wb + o; o += SZ_N128;
  float*    OUTA = Wb + o; o += SZ_N128;
  float*    ASRC = Wb + o; o += SZ_N2;
  float*    ADST = Wb + o; o += SZ_N2;
  float*    AEE  = Wb + o; o += SZ_E2;
  float*    AEN  = Wb + o; o += SZ_N2;
  float*    ALPE = Wb + o; o += SZ_E2;
  float*    ALPN = Wb + o; o += SZ_N2;
  unsigned* MMAX = (unsigned*)(Wb + o); o += SZ_N2;
  float*    SSUM = Wb + o; o += SZ_N2;
  float*    DEG  = Wb + o; o += 10240;
  float*    SB   = Wb + o; o += 1024;     // 8 conv stages x (scale64 + bias64)
  float*    VEC  = Wb + o; o += 768;      // 2 layers x (vsrc,vdst,vedge)[2][64]
  __bf16*   PB   = (__bf16*)(Wb + o); o += PB_TOTAL / 2 + 64;
  if (ws_size < o * sizeof(float)) return;   // insufficient scratch: bail

  // ---- prep: pack weights to bf16 fragment layout ----
  auto pack = [&](const float* Ws, size_t off, int KPER, int Nout, int Npad, int nTap, int mode) {
    int total = nTap * Npad * KPER;
    k_pack_w<<<dim3(cdiv(total, 256)), dim3(256), 0, stream>>>(Ws, PB + off, KPER, Nout, Npad, nTap, mode);
  };
  pack(PF(P_NFC1_W),  p_nfc1, 128, 64, 64, 1, 0);
  pack(PF(P_NFC2_W),  p_nfc2,  64, 64, 64, 1, 0);
  pack(PF(P_EPRE1_W), p_epre1, 64, 64, 64, 1, 0);
  pack(PF(P_EPRE2_W), p_epre2, 64, 64, 64, 1, 0);
  pack(PF(P_G1_WL),   p_Wl1,   64, 128, 128, 1, 0);
  pack(PF(P_G2_WL),   p_Wl2,   64, 128, 128, 1, 0);
  pack(PF(P_EFC1_W),  p_efc1,  64, 64, 64, 1, 0);
  pack(PF(P_EFC2_W),  p_efc2,  64, 64, 64, 1, 0);
  pack(PF(TE0_W1), p_tE_b0w1, 64, 64, 64, 3, 1);
  pack(PF(TE0_W2), p_tE_b0w2, 64, 64, 64, 3, 1);
  pack(PF(TE1_W1), p_tE_b1w1, 64, 64, 64, 3, 1);
  pack(PF(TE1_W2), p_tE_b1w2, 64, 64, 64, 3, 1);
  pack(PF(TN0_W1), p_tN_b0w1, 64, 64, 64, 3, 1);
  pack(PF(TN0_W2), p_tN_b0w2, 64, 64, 64, 3, 1);
  pack(PF(TN1_W1), p_tN_b1w1, 64, 64, 64, 3, 1);
  pack(PF(TN1_W2), p_tN_b1w2, 64, 64, 64, 3, 1);
  pack(PF(P_CLS1_W), p_cls1,  64, 128, 128, 3, 0);
  pack(PF(P_CLS2_W), p_cls2, 128,   8,  16, 1, 0);

  // fold conv bias + BN into scale/bias: slots 0..3 edge TCN, 4..7 node TCN
  auto fold = [&](int gi, int bbi, int bci, int slot) {
    k_fold<<<1, 64, 0, stream>>>(PF(gi), PF(bbi), PF(bci), SB + slot * 128, SB + slot * 128 + 64, 64);
  };
  fold(TE0_BN1G, TE0_BN1B, TE0_B1, 0); fold(TE0_BN2G, TE0_BN2B, TE0_B2, 1);
  fold(TE1_BN1G, TE1_BN1B, TE1_B1, 2); fold(TE1_BN2G, TE1_BN2B, TE1_B2, 3);
  fold(TN0_BN1G, TN0_BN1B, TN0_B1, 4); fold(TN0_BN2G, TN0_BN2B, TN0_B2, 5);
  fold(TN1_BN1G, TN1_BN1B, TN1_B1, 6); fold(TN1_BN2G, TN1_BN2B, TN1_B2, 7);

  auto fillf = [&](float* p, float v, size_t n) { k_fill_f<<<dim3(2048), 256, 0, stream>>>(p, v, n); };
  auto fillu = [&](unsigned* p, unsigned v, size_t n) { k_fill_u<<<dim3(2048), 256, 0, stream>>>(p, v, n); };

  // plain GEMM (T folded into M); abf/obf pick bf16 operand paths
  auto gemm_flat = [&](const void* A, size_t pW, const float* sc, const float* bi,
                       const float* res, void* out, int M, int K, int Nout, int Npad,
                       int relu, int abf, int obf) {
    GP p = { A, nullptr, nullptr, nullptr, nullptr, nullptr,
             M, M, M, 0, 0, 0, 1, K, PB + pW, sc, bi, res, out,
             M, Nout, Npad, relu, 1, abf, obf };
    run_gemm(stream, p);
  };
  // causal conv (3-tap shifted GEMM over grid.y = T)
  auto conv = [&](const void* In, void* Outp, const float* Resp, int Mt, int dil,
                  size_t pW, int slot, int abf, int obf) {
    GP p = { In, In, In, nullptr, nullptr, nullptr,
             Mt, Mt, Mt, -2 * dil, -dil, 0, 3, 64, PB + pW,
             SB + slot * 128, SB + slot * 128 + 64, Resp, Outp,
             Mt, 64, 64, 1, T_, abf, obf };
    run_gemm(stream, p);
  };

  // ---- per-timestep encoders: chain temps in bf16 (halved traffic) ----
  gemm_flat(X,   p_nfc1,  nullptr, PF(P_NFC1_B),  nullptr, HB0, T_ * NN, 128, 64, 64, 1, 0, 1);
  gemm_flat(HB0, p_nfc2,  nullptr, PF(P_NFC2_B),  nullptr, HN0, T_ * NN,  64, 64, 64, 1, 1, 0);
  gemm_flat(EAT, p_epre1, nullptr, PF(P_EPRE1_B), nullptr, EB0, T_ * EE,  64, 64, 64, 1, 0, 1);
  gemm_flat(EB0, p_epre2, nullptr, PF(P_EPRE2_B), nullptr, EA0, T_ * EE,  64, 64, 64, 1, 1, 0);

  // self-loop edge attr = per-node mean of incoming edge attrs (shared by both GATs)
  fillf(DEG, 0.f, NN);
  fillf(LOOP, 0.f, SZ_N64);
  k_deg<<<dim3(cdiv(EE, 256)), 256, 0, stream>>>(DST, DEG);
  k_loopsum<<<dim3(cdiv((size_t)T_ * EE * 64, 256)), 256, 0, stream>>>(EA0, DST, LOOP);
  k_loopdiv<<<dim3(cdiv(SZ_N64, 256)), 256, 0, stream>>>(LOOP, DEG);

  // ---- GAT layer (all f32: scatter/softmax path) ----
  auto gat_layer = [&](const float* Hin, float* Hout, int iWl, int iWe, int iASRCp,
                       int iADSTp, int iAEDGEp, int iBIAS, int iBNB, int iBNG,
                       size_t pWl, float* vecs) {
    float* vsrc = vecs, *vdst = vecs + 128, *vedge = vecs + 256;
    k_attvec<<<1, 128, 0, stream>>>(PF(iWl), PF(iASRCp), vsrc);
    k_attvec<<<1, 128, 0, stream>>>(PF(iWl), PF(iADSTp), vdst);
    k_attvec<<<1, 128, 0, stream>>>(PF(iWe), PF(iAEDGEp), vedge);
    gemm_flat(Hin, pWl, nullptr, nullptr, nullptr, XL, T_ * NN, 64, 128, 128, 0, 0, 0); // xl
    k_nodeatt<<<dim3(cdiv((size_t)T_ * NN, 256)), 256, 0, stream>>>(Hin, vsrc, vdst, ASRC, ADST, (size_t)T_ * NN);
    k_dot2<<<dim3(cdiv((size_t)T_ * EE, 256)), 256, 0, stream>>>(EA0, vedge, AEE, (size_t)T_ * EE);
    k_dot2<<<dim3(cdiv((size_t)T_ * NN, 256)), 256, 0, stream>>>(LOOP, vedge, AEN, (size_t)T_ * NN);
    fillu(MMAX, ENC_NEG_INF, SZ_N2);
    k_logit_e<<<dim3(cdiv(SZ_E2, 256)), 256, 0, stream>>>(ASRC, ADST, AEE, SRC, DST, ALPE, MMAX);
    k_logit_n<<<dim3(cdiv(SZ_N2, 256)), 256, 0, stream>>>(ASRC, ADST, AEN, ALPN, MMAX);
    fillf(SSUM, 0.f, SZ_N2);
    k_exps_e<<<dim3(cdiv(SZ_E2, 256)), 256, 0, stream>>>(ALPE, MMAX, SSUM, DST);
    k_exps_n<<<dim3(cdiv(SZ_N2, 256)), 256, 0, stream>>>(ALPN, MMAX, SSUM);
    fillf(OUTA, 0.f, SZ_N128);
    k_scat_e<<<dim3(cdiv((size_t)T_ * EE * 64, 256)), 256, 0, stream>>>(ALPE, SSUM, XL, SRC, DST, OUTA);
    k_scat_n<<<dim3(cdiv(SZ_N64, 256)), 256, 0, stream>>>(ALPN, SSUM, XL, OUTA);
    k_gat_final<<<dim3(cdiv(SZ_N64, 256)), 256, 0, stream>>>(OUTA, PF(iBIAS), PF(iBNG), PF(iBNB), Hout);
  };
  gat_layer(HN0, HN1, P_G1_WL, P_G1_WE, P_G1_ASRC, P_G1_ADST, P_G1_AEDGE, P_G1_BIAS,
            P_BN1_B, P_BN1_G, p_Wl1, VEC);
  gat_layer(HN1, HN2, P_G2_WL, P_G2_WE, P_G2_ASRC, P_G2_ADST, P_G2_AEDGE, P_G2_BIAS,
            P_BN2_B, P_BN2_G, p_Wl2, VEC + 384);

  // ---- edge MLP after GATs (bf16 temp) ----
  gemm_flat(EA0, p_efc1, nullptr, PF(P_EFC1_B), nullptr, EB0, T_ * EE, 64, 64, 64, 1, 0, 1);
  gemm_flat(EB0, p_efc2, nullptr, PF(P_EFC2_B), nullptr, EA1, T_ * EE, 64, 64, 64, 1, 1, 0);

  // ---- TCNs: causal K=3 convs as 3-tap shifted WMMA GEMMs ----
  // conv1 outputs feed only conv2 -> bf16; conv2 outputs carry residuals -> f32
  conv(EA1, EB0, nullptr, EE, 1, p_tE_b0w1, 0, 0, 1);   // edge b0 conv1
  conv(EB0, EA2, EA1,     EE, 1, p_tE_b0w2, 1, 1, 0);   //         conv2 + res -> y0
  conv(EA2, EB0, nullptr, EE, 2, p_tE_b1w1, 2, 0, 1);   // edge b1 (dil 2)
  conv(EB0, EA1, EA2,     EE, 2, p_tE_b1w2, 3, 1, 0);   // te -> EA1
  conv(HN2, HB0, nullptr, NN, 1, p_tN_b0w1, 4, 0, 1);   // node b0
  conv(HB0, HN1, HN2,     NN, 1, p_tN_b0w2, 5, 1, 0);   // y0 -> HN1
  conv(HN1, HB0, nullptr, NN, 2, p_tN_b1w1, 6, 0, 1);   // node b1
  conv(HB0, HN0, HN1,     NN, 2, p_tN_b1w2, 7, 1, 0);   // tn -> HN0

  // ---- classifier: cls1 gathers [tn[src]|tn[dst]|te], writes bf16 hidden ----
  __bf16* H1 = (__bf16*)EA0;  // EA0 region (dead) = [T,E,128] bf16 exactly
  {
    GP p = { HN0, HN0, EA1, SRC, DST, nullptr, NN, NN, EE, 0, 0, 0,
             3, 64, PB + p_cls1, nullptr, PF(P_CLS1_B), nullptr, H1,
             EE, 128, 128, 1, T_, 0, 1 };
    run_gemm(stream, p);
  }
  gemm_flat(H1, p_cls2, nullptr, PF(P_CLS2_B), nullptr, d_out,
            T_ * EE, 128, 8, 16, 0, 1, 0);   // logits [T,E,8] f32
  #undef PF
}
